// graphNetwork_18451179503910
// MI455X (gfx1250) — matrix operand order repaired and checked by reading, not compiled, not run on hardware
//
#include <hip/hip_runtime.h>
#include <stdint.h>
#include <stddef.h>
#include <float.h>

typedef __attribute__((ext_vector_type(16))) __bf16 v16bf;
typedef __attribute__((ext_vector_type(8)))  float  v8f;

#define GN_H   0.1f
#define GN_EPS 1e-5f

// ---------------------------------------------------------------------------
// small utility kernels
// ---------------------------------------------------------------------------

// Pack W (64 x K, f32 row-major) into WMMA A-fragment order, bf16:
// dword index = ((kstep*4 + mt)*32 + lane)*8 + r  holds {W[m][k], W[m][k+1]}
// with m = mt*16 + lane%16, k = kstep*32 + 2r + 16*(lane/16)
__global__ void k_pack_w(const float* __restrict__ s, __bf16* __restrict__ d, int K) {
  int t = blockIdx.x * blockDim.x + threadIdx.x;  // one thread per dword pair
  int n = 32 * K;
  if (t >= n) return;
  int r    = t & 7;
  int lane = (t >> 3) & 31;
  int mt   = (t >> 8) & 3;
  int ks   = t >> 10;
  int m    = mt * 16 + (lane & 15);
  int half = lane >> 4;
  int kl   = 2 * r + 16 * half;
  int k    = ks * 32 + kl;
  d[2 * t]     = (__bf16)s[m * K + k];
  d[2 * t + 1] = (__bf16)s[m * K + k + 1];
}

__global__ void k_zero(float* __restrict__ p, int n) {
  int t = blockIdx.x * blockDim.x + threadIdx.x;
  if (t < n) p[t] = 0.f;
}

// acc = {sum, sumsq}  ->  out = {mean, rstd}
__global__ void k_finalize_stats(const float* __restrict__ acc,
                                 float* __restrict__ out, float invCount) {
  float m = acc[0] * invCount;
  float v = acc[1] * invCount - m * m;
  out[0] = m;
  out[1] = rsqrtf(v + GN_EPS);
}

// catn rows 0..127 <- 0 ; rows 128..191 <- xn
__global__ void k_init_catn(const float* __restrict__ xn, float* __restrict__ catn, int N) {
  size_t t = (size_t)blockIdx.x * blockDim.x + threadIdx.x;
  size_t total = (size_t)192 * N;
  if (t >= total) return;
  size_t k = t / (size_t)N;
  catn[t] = (k < 128) ? 0.f : xn[t - (size_t)128 * N];
}

// scatter-add edges into catn: rows 0..63 = aveE = 0.5*(si+sj), rows 64..127 = divE = si-sj
__global__ void k_scatter(const float* __restrict__ xe, const int* __restrict__ iInd,
                          const int* __restrict__ jInd, float* __restrict__ catn,
                          int E, int N) {
  size_t t = (size_t)blockIdx.x * blockDim.x + threadIdx.x;
  if (t >= (size_t)64 * E) return;
  int k = (int)(t / (size_t)E);
  int e = (int)(t - (size_t)k * E);
  float v = xe[t];
  int i = iInd[e], j = jInd[e];
  atomicAdd(&catn[(size_t)k * N + i], 0.5f * v);
  atomicAdd(&catn[(size_t)k * N + j], 0.5f * v);
  atomicAdd(&catn[(size_t)(64 + k) * N + i], v);
  atomicAdd(&catn[(size_t)(64 + k) * N + j], -v);
}

// xe_out = xe + H * ln(dxe)   (ln via precomputed scalar mean/rstd)
__global__ void k_edge_update(const float* __restrict__ xe, const float* __restrict__ dxe,
                              const float* __restrict__ st, float* __restrict__ outp,
                              size_t n) {
  size_t t = (size_t)blockIdx.x * blockDim.x + threadIdx.x;
  if (t >= n) return;
  outp[t] = xe[t] + GN_H * ((dxe[t] - st[0]) * st[1]);
}

// ---------------------------------------------------------------------------
// Fused WMMA GEMM:  Out(64 x ncols) = alpha * Wp(64 x K, packed bf16) * B(K x ncols) [+ Resid]
//   - Packed weights staged in LDS once per block; A fragments are 32B ds loads
//     with compile-time offsets (K is a template parameter, k-loop fully unrolled).
//   - B source: plain row-major f32 matrix (optionally global-LN-normalized + ReLU,
//               applied branchlessly), or on-the-fly edge-concat
//               [0.5*(xn_i+xn_j) ; xe ; xn_i - xn_j]; region per k-step is an
//               unroll-time constant (ks>>6), so coefficients/pointers const-fold.
//   - optional fused accumulation of sum/sumsq of the written outputs (global LN)
// One wave handles 16 columns and all 64 output rows (4 WMMA M-tiles).
// ncols must be a multiple of 16 (N=50000, E=400000 both are).
// ---------------------------------------------------------------------------
template <int K, bool EDGECAT>
__global__ __launch_bounds__(128)
void k_gemm64(const __bf16* __restrict__ Wpacked,
              const float* __restrict__ Bplain,
              const float* __restrict__ gxn, const float* __restrict__ gxe,
              const int* __restrict__ iInd, const int* __restrict__ jInd, int Nn,
              const float* __restrict__ prenorm, int relu,
              float* __restrict__ Out, int ncols,
              float alpha, const float* __restrict__ Resid,
              float* __restrict__ statsAcc) {
  __shared__ __bf16 sW[K * 64];   // packed A fragments (K/32 * 4 tiles * 32 lanes * 32B)
  __shared__ float  sred[2];

  // ---- cooperative global -> LDS stage of the packed weights ----
  {
    const uint4* src = (const uint4*)Wpacked;
    uint4*       dst = (uint4*)sW;
#pragma unroll
    for (int it = 0; it < (K * 8) / 128; ++it)
      dst[threadIdx.x + it * 128] = src[threadIdx.x + it * 128];
  }
  __syncthreads();

  const int lane = threadIdx.x & 31;
  const int wave = threadIdx.x >> 5;
  const int col0 = (blockIdx.x * 4 + wave) * 16;
  const int n    = lane & 15;
  const int half = lane >> 4;

  if (col0 < ncols) {
    const int col = col0 + n;
    float mean = 0.f, rstd = 1.f, neglim = -FLT_MAX;
    if (prenorm) { mean = prenorm[0]; rstd = prenorm[1]; if (relu) neglim = 0.f; }
    int ii = 0, jj = 0;
    if (EDGECAT) { ii = iInd[col]; jj = jInd[col]; }

    const v16bf* __restrict__ WpL = (const v16bf*)sW;

    v8f acc[4];
#pragma unroll
    for (int mt = 0; mt < 4; ++mt) acc[mt] = (v8f){0.f,0.f,0.f,0.f,0.f,0.f,0.f,0.f};

    const int kb0 = half * 16;
#pragma unroll
    for (int ks = 0; ks < K; ks += 32) {
      const int kb = ks + kb0;
      // ---- B tile (32 x 16, bf16): lane = column, element e <-> k = kb + e ----
      v16bf b;
      if (!EDGECAT) {
        const char* p = (const char*)(Bplain + (size_t)kb * ncols + col);
        const size_t sb = (size_t)ncols * sizeof(float);
#pragma unroll
        for (int e = 0; e < 16; ++e) {
          float v = *(const float*)p;
          v = fmaxf((v - mean) * rstd, neglim);
          b[e] = (__bf16)v;
          p += sb;
        }
      } else {
        // region of this k-step (uniform: ks%64 in {0,32}, +16*half never wraps)
        const int reg = ks >> 6;              // 0=intX, 1=xe, 2=gradX (unroll const)
        const int kr  = kb & 63;
        const float* a0 = (reg == 1) ? (gxe + (size_t)kr * ncols + col)
                                     : (gxn + (size_t)kr * Nn + ii);
        const float* b0 = (reg == 1) ? (gxe + (size_t)kr * ncols + col)
                                     : (gxn + (size_t)kr * Nn + jj);
        const size_t sb = ((reg == 1) ? (size_t)ncols : (size_t)Nn) * sizeof(float);
        const float ca = (reg == 2) ? 1.0f : 0.5f;
        const float cb = (reg == 2) ? -1.0f : 0.5f;
        const char* pa = (const char*)a0;
        const char* pb = (const char*)b0;
#pragma unroll
        for (int e = 0; e < 16; ++e) {
          float v = ca * (*(const float*)pa) + cb * (*(const float*)pb);
          b[e] = (__bf16)v;
          pa += sb; pb += sb;
        }
      }
      // ---- 4 M-tiles: A fragments from LDS (compile-time offsets), then WMMAs ----
      const int ks32 = ks >> 5;
      v16bf a[4];
#pragma unroll
      for (int mt = 0; mt < 4; ++mt)
        a[mt] = WpL[(ks32 * 4 + mt) * 32 + lane];
#pragma unroll
      for (int mt = 0; mt < 4; ++mt)
        acc[mt] = __builtin_amdgcn_wmma_f32_16x16x32_bf16(
            /*neg_a=*/false, a[mt], /*neg_b=*/false, b,
            /*c_mod=*/(short)0, acc[mt], /*reuse_a=*/false, /*reuse_b=*/false);
    }

    // ---- epilogue: scale, residual, store, local LN stats ----
    float ls = 0.f, lss = 0.f;
#pragma unroll
    for (int mt = 0; mt < 4; ++mt) {
#pragma unroll
      for (int r = 0; r < 8; ++r) {
        const int m = mt * 16 + r + 8 * half;
        float v = alpha * acc[mt][r];
        if (Resid) v += Resid[(size_t)m * ncols + col];
        Out[(size_t)m * ncols + col] = v;
        ls += v; lss += v * v;
      }
    }
    if (statsAcc) {
      if (threadIdx.x == 0) { sred[0] = 0.f; sred[1] = 0.f; }
      __syncthreads();
      atomicAdd(&sred[0], ls);
      atomicAdd(&sred[1], lss);
      __syncthreads();
      if (threadIdx.x == 0) {
        atomicAdd(&statsAcc[0], sred[0]);
        atomicAdd(&statsAcc[1], sred[1]);
      }
    }
  }
}

// ---------------------------------------------------------------------------
// host orchestration
// ---------------------------------------------------------------------------
extern "C" void kernel_launch(void* const* d_in, const int* in_sizes, int n_in,
                              void* d_out, int out_size, void* d_ws, size_t ws_size,
                              hipStream_t stream) {
  (void)n_in; (void)out_size; (void)ws_size;
  const float* xn_in = (const float*)d_in[0];
  const float* xe_in = (const float*)d_in[1];
  const int*   iInd  = (const int*)d_in[2];
  const int*   jInd  = (const int*)d_in[3];
  const float* K1N = (const float*)d_in[4];
  const float* K2N = (const float*)d_in[5];
  const float* K1E = (const float*)d_in[6];
  const float* K2E = (const float*)d_in[7];
  const float* KNc = (const float*)d_in[8];
  const float* KE1 = (const float*)d_in[9];    // (2, 64, 192)
  const float* KE2 = (const float*)d_in[10];   // (2, 64, 64)
  const float* KN1 = (const float*)d_in[11];   // (2, 64, 192)
  const float* KN2 = (const float*)d_in[12];   // (2, 64, 64)

  const int N = in_sizes[0] / 64;
  const int E = in_sizes[1] / 64;

  // ---- carve workspace ----
  char* base = (char*)d_ws;
  auto carve = [&](size_t bytes) -> char* {
    char* p = base;
    base += (bytes + 255) & ~(size_t)255;
    return p;
  };
  float*  statsAcc = (float*)carve(32 * sizeof(float));   // 16 slots x {sum,sumsq}
  float*  statsOut = (float*)carve(32 * sizeof(float));   // 16 slots x {mean,rstd}
  __bf16* wbf      = (__bf16*)carve(86016 * sizeof(__bf16));
  float*  tmp_n    = (float*)carve((size_t)64 * N * sizeof(float));
  float*  xn_a     = (float*)carve((size_t)64 * N * sizeof(float));
  float*  xn_b     = (float*)carve((size_t)64 * N * sizeof(float));
  float*  catn     = (float*)carve((size_t)192 * N * sizeof(float));
  float*  tmp_e    = (float*)carve((size_t)64 * E * sizeof(float));
  float*  xe_a     = (float*)carve((size_t)64 * E * sizeof(float));
  float*  xe_b     = (float*)carve((size_t)64 * E * sizeof(float));

  __bf16* wK1N = wbf;                // 64x64  -> 4096
  __bf16* wK2N = wK1N + 4096;
  __bf16* wK1E = wK2N + 4096;
  __bf16* wK2E = wK1E + 4096;
  __bf16* wKNc = wK2E + 4096;
  __bf16* wKE1 = wKNc + 4096;        // 2 * 12288
  __bf16* wKE2 = wKE1 + 2 * 12288;   // 2 * 4096
  __bf16* wKN1 = wKE2 + 2 * 4096;    // 2 * 12288
  __bf16* wKN2 = wKN1 + 2 * 12288;   // 2 * 4096

  auto pack = [&](const float* s, __bf16* d, int K) {
    int nthr = 32 * K;  // one thread per packed dword
    k_pack_w<<<(nthr + 255) / 256, 256, 0, stream>>>(s, d, K);
  };
  pack(K1N, wK1N, 64); pack(K2N, wK2N, 64);
  pack(K1E, wK1E, 64); pack(K2E, wK2E, 64);
  pack(KNc, wKNc, 64);
  for (int l = 0; l < 2; ++l) {
    pack(KE1 + (size_t)l * 12288, wKE1 + (size_t)l * 12288, 192);
    pack(KE2 + (size_t)l * 4096,  wKE2 + (size_t)l * 4096,  64);
    pack(KN1 + (size_t)l * 12288, wKN1 + (size_t)l * 12288, 192);
    pack(KN2 + (size_t)l * 4096,  wKN2 + (size_t)l * 4096,  64);
  }

  k_zero<<<1, 64, 0, stream>>>(statsAcc, 32);

  auto gemm = [&](const __bf16* W, int K, const float* B,
                  const float* gxn, const float* gxe, int edgecat,
                  const float* prenorm, int relu,
                  float* Out, int ncols, float alpha, const float* Resid,
                  float* sAcc) {
    dim3 grid((unsigned)((ncols / 16 + 3) / 4));
    if (edgecat)
      k_gemm64<192, true><<<grid, 128, 0, stream>>>(W, B, gxn, gxe, iInd, jInd, N,
                                                    prenorm, relu, Out, ncols, alpha, Resid, sAcc);
    else if (K == 192)
      k_gemm64<192, false><<<grid, 128, 0, stream>>>(W, B, gxn, gxe, iInd, jInd, N,
                                                     prenorm, relu, Out, ncols, alpha, Resid, sAcc);
    else
      k_gemm64<64, false><<<grid, 128, 0, stream>>>(W, B, gxn, gxe, iInd, jInd, N,
                                                    prenorm, relu, Out, ncols, alpha, Resid, sAcc);
  };
  auto fin = [&](int slot, size_t count) {
    k_finalize_stats<<<1, 1, 0, stream>>>(statsAcc + 2 * slot, statsOut + 2 * slot,
                                          1.0f / (float)count);
  };

  int slot = 0;
  // ---- node open: xn = K2N x relu(ln(K1N x xn)) ----
  int sN = slot++;
  gemm(wK1N, 64, xn_in, nullptr, nullptr, 0, nullptr, 0, tmp_n, N, 1.f, nullptr, statsAcc + 2 * sN);
  fin(sN, (size_t)64 * N);
  gemm(wK2N, 64, tmp_n, nullptr, nullptr, 0, statsOut + 2 * sN, 1, xn_a, N, 1.f, nullptr, nullptr);

  // ---- edge open: xe = K2E x relu(ln(K1E x xe)) ----
  int sE = slot++;
  gemm(wK1E, 64, xe_in, nullptr, nullptr, 0, nullptr, 0, tmp_e, E, 1.f, nullptr, statsAcc + 2 * sE);
  fin(sE, (size_t)64 * E);
  gemm(wK2E, 64, tmp_e, nullptr, nullptr, 0, statsOut + 2 * sE, 1, xe_a, E, 1.f, nullptr, nullptr);

  float* xn_cur = xn_a; float* xn_nxt = xn_b;
  float* xe_cur = xe_a; float* xe_nxt = xe_b;
  float* out_xn = (float*)d_out;
  float* out_xe = (float*)d_out + (size_t)64 * N;

  for (int l = 0; l < 2; ++l) {
    int sDE0 = slot++, sDE1 = slot++, sDN0 = slot++;
    // dxe0 = KE1[l] x concat(intX, xe, gradX)  -- concat synthesized in B loader
    gemm(wKE1 + (size_t)l * 12288, 192, nullptr, xn_cur, xe_cur, 1,
         nullptr, 0, tmp_e, E, 1.f, nullptr, statsAcc + 2 * sDE0);
    fin(sDE0, (size_t)64 * E);
    // dxe1 = KE2[l] x relu(ln(dxe0))   (in-place: wave reads its columns before storing)
    gemm(wKE2 + (size_t)l * 4096, 64, tmp_e, nullptr, nullptr, 0,
         statsOut + 2 * sDE0, 1, tmp_e, E, 1.f, nullptr, statsAcc + 2 * sDE1);
    fin(sDE1, (size_t)64 * E);
    // catn = [aveE ; divE ; xn]  via zero+copy then scatter-add (uses OLD xe)
    k_init_catn<<<(unsigned)(((size_t)192 * N + 255) / 256), 256, 0, stream>>>(xn_cur, catn, N);
    k_scatter<<<(unsigned)(((size_t)64 * E + 255) / 256), 256, 0, stream>>>(xe_cur, iInd, jInd, catn, E, N);
    // dxn0 = KN1[l] x catn
    gemm(wKN1 + (size_t)l * 12288, 192, catn, nullptr, nullptr, 0,
         nullptr, 0, tmp_n, N, 1.f, nullptr, statsAcc + 2 * sDN0);
    fin(sDN0, (size_t)64 * N);
    // xn_next = xn_cur + H * (KN2[l] x relu(ln(dxn0)))   (residual fused)
    gemm(wKN2 + (size_t)l * 4096, 64, tmp_n, nullptr, nullptr, 0,
         statsOut + 2 * sDN0, 1, xn_nxt, N, GN_H, xn_cur, nullptr);
    // xe_next = xe_cur + H * ln(dxe1)  (last layer writes straight to output)
    float* xedst = (l == 1) ? out_xe : xe_nxt;
    k_edge_update<<<(unsigned)(((size_t)64 * E + 255) / 256), 256, 0, stream>>>(
        xe_cur, tmp_e, statsOut + 2 * sDE1, xedst, (size_t)64 * E);
    // swap
    float* t = xn_cur; xn_cur = xn_nxt; xn_nxt = t;
    if (l == 0) { t = xe_cur; xe_cur = xe_nxt; xe_nxt = t; }
  }

  // ---- close: out_xn = KNclose x xn ----
  gemm(wKNc, 64, xn_cur, nullptr, nullptr, 0, nullptr, 0, out_xn, N, 1.f, nullptr, nullptr);
}
